// CausalSelfAttention_48945447305683
// MI455X (gfx1250) — matrix operand order repaired and checked by reading, not compiled, not run on hardware
//
#include <hip/hip_runtime.h>
#include <cstdint>
#include <cstddef>

// ---- CDNA5 wave32 WMMA types ----
typedef __attribute__((ext_vector_type(16))) __bf16 v16bf;
typedef __attribute__((ext_vector_type(8)))  __bf16 v8bf;
typedef __attribute__((ext_vector_type(8)))  float  v8f;

// Problem constants (B=4, T=2048, C=1024, H=16, D=64)
#define NBATCH 4
#define TSEQ   2048
#define CEMB   1024
#define NHEAD  16
#define HDIM   64
#define MROWS  (NBATCH * TSEQ)   // 8192
#define NBH    (NBATCH * NHEAD)  // 64

union V16U { v16bf v; v8bf h[2]; };

// A-fragment (16x32 bf16, M=lane%16, K contiguous by lane-half, matching B):
// lane half hf: elements 0..15 = K hf*16+0..15
__device__ inline v16bf load_frag_a(const __bf16* rowp, int hf) {
  V16U u;
  u.h[0] = *(const v8bf*)(rowp + hf * 16);
  u.h[1] = *(const v8bf*)(rowp + hf * 16 + 8);
  return u.v;
}
// B-fragment (32x16 bf16, N=lane%16, lanes 0-15 K=0..15, lanes 16-31 K=16..31):
// caller passes pointer to the 16 contiguous K values for this lane.
__device__ inline v16bf load_frag_b(const __bf16* p) {
  V16U u;
  u.h[0] = *(const v8bf*)(p);
  u.h[1] = *(const v8bf*)(p + 8);
  return u.v;
}
__device__ inline v8f wmma_bf16(v16bf a, v16bf b, v8f c) {
  return __builtin_amdgcn_wmma_f32_16x16x32_bf16(false, a, false, b, (short)0, c,
                                                 false, false);
}

// Row reductions across the 16-lane half-groups (C-layout: cols across lanes)
__device__ inline float rmax16(float v) {
  v = fmaxf(v, __shfl_xor(v, 1, 32));
  v = fmaxf(v, __shfl_xor(v, 2, 32));
  v = fmaxf(v, __shfl_xor(v, 4, 32));
  v = fmaxf(v, __shfl_xor(v, 8, 32));
  return v;
}
__device__ inline float rsum16(float v) {
  v += __shfl_xor(v, 1, 32);
  v += __shfl_xor(v, 2, 32);
  v += __shfl_xor(v, 4, 32);
  v += __shfl_xor(v, 8, 32);
  return v;
}

// ---------------- bf16 conversion kernels ----------------
__global__ void k_cvt(const float* __restrict__ in, __bf16* __restrict__ out, int n) {
  int i = blockIdx.x * blockDim.x + threadIdx.x;
  int stride = gridDim.x * blockDim.x;
  for (; i < n; i += stride) out[i] = (__bf16)in[i];
}
// in[K][N] (row-major) -> out[N][K] bf16 (transposed so B-fragments load contiguous)
__global__ void k_cvt_t(const float* __restrict__ in, __bf16* __restrict__ out,
                        int K, int N) {
  long total = (long)K * N;
  long i = (long)blockIdx.x * blockDim.x + threadIdx.x;
  long stride = (long)gridDim.x * blockDim.x;
  for (; i < total; i += stride) {
    long k = i % K, n = i / K;
    out[n * (long)K + k] = (__bf16)in[k * (long)N + n];
  }
}

// ---------------- shared GEMM mainloop: one wave -> 16(M) x 64(N) tile ----------------
// A: MxK row-major bf16 (lda=K), Bt: NxK row-major bf16 (ldb=K)
__device__ inline void gemm_wave_16x64(const __bf16* __restrict__ A,
                                       const __bf16* __restrict__ Bt,
                                       int KDIM, int m0, int n0, v8f acc[4]) {
  const int lane = threadIdx.x & 31;
  const int hf = lane >> 4;
  const int cl = lane & 15;
  const __bf16* arow = A + (size_t)(m0 + cl) * KDIM;
  for (int k0 = 0; k0 < KDIM; k0 += 32) {
    v16bf a = load_frag_a(arow + k0, hf);
#pragma unroll
    for (int i = 0; i < 4; ++i) {
      const __bf16* bp = Bt + (size_t)(n0 + i * 16 + cl) * KDIM + k0 + hf * 16;
      acc[i] = wmma_bf16(a, load_frag_b(bp), acc[i]);
    }
  }
}

// ---------------- QKV projection: qkv = xb @ Wqkv + b; scatter to Q/K/Vt bf16 ----------------
__global__ __launch_bounds__(128) void k_qkv_gemm(const __bf16* __restrict__ xb,
                                                  const __bf16* __restrict__ Wt,
                                                  const float* __restrict__ bias,
                                                  __bf16* __restrict__ Qo,
                                                  __bf16* __restrict__ Ko,
                                                  __bf16* __restrict__ Vo) {
  const int w = threadIdx.x >> 5;
  const int lane = threadIdx.x & 31;
  const int hf = lane >> 4;
  const int cl = lane & 15;
  const int m0 = blockIdx.y * 64 + w * 16;
  const int n0 = blockIdx.x * 64;
  v8f acc[4];
#pragma unroll
  for (int i = 0; i < 4; ++i)
#pragma unroll
    for (int r = 0; r < 8; ++r) acc[i][r] = 0.f;

  gemm_wave_16x64(xb, Wt, CEMB, m0, n0, acc);

#pragma unroll
  for (int i = 0; i < 4; ++i) {
    const int gn = n0 + i * 16 + cl;           // [0,3072)
    const int sec = gn >> 10;                  // 0=Q 1=K 2=V (uniform per i)
    const int c = gn & (CEMB - 1);
    const int h = c >> 6, d = c & 63;
    const float bv = bias[gn];
#pragma unroll
    for (int r = 0; r < 8; ++r) {
      const int gm = m0 + r + 8 * hf;          // [0,8192)
      const int b = gm >> 11, t = gm & (TSEQ - 1);
      const size_t bh = (size_t)(b * NHEAD + h);
      const __bf16 val = (__bf16)(acc[i][r] + bv);
      if (sec == 0)      Qo[(bh * TSEQ + t) * HDIM + d] = val;   // [B,H,T,D]
      else if (sec == 1) Ko[(bh * TSEQ + t) * HDIM + d] = val;   // [B,H,T,D]
      else               Vo[(bh * HDIM + d) * TSEQ + t] = val;   // [B,H,D,T] (transposed)
    }
  }
}

// ---------------- flash attention: one wave = one 16-query tile ----------------
__global__ __launch_bounds__(128) void k_attn(const __bf16* __restrict__ Q,
                                              const __bf16* __restrict__ Kc,
                                              const __bf16* __restrict__ Vt,
                                              __bf16* __restrict__ AO) {
  __shared__ __bf16 Pl[4][16 * 32];   // per-wave P tile for C-layout -> A-fragment relayout
  const int w = threadIdx.x >> 5;
  const int lane = threadIdx.x & 31;
  const int hf = lane >> 4;
  const int cl = lane & 15;
  const int tile = blockIdx.x * 4 + w;          // 8192 tiles total
  const int bh = tile >> 7;                     // 128 query tiles per (b,h)
  const int q0 = (tile & 127) << 4;

  const __bf16* Qb = Q + (size_t)bh * TSEQ * HDIM;
  const __bf16* Kb = Kc + (size_t)bh * TSEQ * HDIM;
  const __bf16* Vb = Vt + (size_t)bh * HDIM * TSEQ;

  // Q fragments for D=64 (two K=32 chunks), reused for all key blocks
  const __bf16* qrow = Qb + (size_t)(q0 + cl) * HDIM;
  const v16bf qa0 = load_frag_a(qrow, hf);       // d 0..31
  const v16bf qa1 = load_frag_a(qrow + 32, hf);  // d 32..63

  v8f m_r, l_r, o0, o1, o2, o3;
#pragma unroll
  for (int r = 0; r < 8; ++r) {
    m_r[r] = -1e30f; l_r[r] = 0.f;
    o0[r] = 0.f; o1[r] = 0.f; o2[r] = 0.f; o3[r] = 0.f;
  }
  const float SCALE = 0.18033688011112042f;      // log2(e)/sqrt(64)

  for (int k0 = 0; k0 <= q0 + 15; k0 += 32) {
    // S = Q K^T : two 16x16 tiles (keys k0..+15, k0+16..+31), K stored [T,D] so
    // B-fragment K-dim (=d) is contiguous per lane.
    v8f s0, s1;
#pragma unroll
    for (int r = 0; r < 8; ++r) { s0[r] = 0.f; s1[r] = 0.f; }
    {
      const __bf16* kr = Kb + (size_t)(k0 + cl) * HDIM + hf * 16;
      s0 = wmma_bf16(qa0, load_frag_b(kr), s0);
      s0 = wmma_bf16(qa1, load_frag_b(kr + 32), s0);
    }
    {
      const __bf16* kr = Kb + (size_t)(k0 + 16 + cl) * HDIM + hf * 16;
      s1 = wmma_bf16(qa0, load_frag_b(kr), s1);
      s1 = wmma_bf16(qa1, load_frag_b(kr + 32), s1);
    }
    // scale + causal mask + online softmax (rows live in VGPR index, cols across lanes)
    v8f p0, p1;
#pragma unroll
    for (int r = 0; r < 8; ++r) {
      const int qg = q0 + r + 8 * hf;
      const float a0 = (k0 + cl <= qg)      ? s0[r] * SCALE : -1e30f;
      const float a1 = (k0 + 16 + cl <= qg) ? s1[r] * SCALE : -1e30f;
      const float mx = rmax16(fmaxf(a0, a1));
      const float mn = fmaxf(m_r[r], mx);
      const float al = exp2f(m_r[r] - mn);
      p0[r] = exp2f(a0 - mn);
      p1[r] = exp2f(a1 - mn);
      l_r[r] = l_r[r] * al + rsum16(p0[r] + p1[r]);
      m_r[r] = mn;
      o0[r] *= al; o1[r] *= al; o2[r] *= al; o3[r] *= al;
    }
    // P: C-layout -> LDS -> A-fragment (wave-local DS ops are in-order, no barrier)
    __bf16* pw = &Pl[w][0];
#pragma unroll
    for (int r = 0; r < 8; ++r) {
      const int m = r + 8 * hf;
      pw[m * 32 + cl] = (__bf16)p0[r];
      pw[m * 32 + 16 + cl] = (__bf16)p1[r];
    }
    const v16bf pa = load_frag_a(pw + cl * 32, hf);
    // O += P V : V stored transposed [D,T] so B-fragment K-dim (=key) is contiguous
    o0 = wmma_bf16(pa, load_frag_b(Vb + (size_t)(cl) * TSEQ + k0 + hf * 16), o0);
    o1 = wmma_bf16(pa, load_frag_b(Vb + (size_t)(16 + cl) * TSEQ + k0 + hf * 16), o1);
    o2 = wmma_bf16(pa, load_frag_b(Vb + (size_t)(32 + cl) * TSEQ + k0 + hf * 16), o2);
    o3 = wmma_bf16(pa, load_frag_b(Vb + (size_t)(48 + cl) * TSEQ + k0 + hf * 16), o3);
  }

  // normalize and write attention output as bf16 in [B,T,C] for the out-proj GEMM
  const int b = bh >> 4, h = bh & 15;
#pragma unroll
  for (int r = 0; r < 8; ++r) {
    const int trow = q0 + r + 8 * hf;
    const float inv = 1.f / l_r[r];
    const size_t base = ((size_t)b * TSEQ + trow) * CEMB + h * HDIM;
    AO[base + cl]      = (__bf16)(o0[r] * inv);
    AO[base + 16 + cl] = (__bf16)(o1[r] * inv);
    AO[base + 32 + cl] = (__bf16)(o2[r] * inv);
    AO[base + 48 + cl] = (__bf16)(o3[r] * inv);
  }
}

// ---------------- output projection: out = AO @ Wout + b_out (fp32) ----------------
__global__ __launch_bounds__(128) void k_out_gemm(const __bf16* __restrict__ AO,
                                                  const __bf16* __restrict__ Wt,
                                                  const float* __restrict__ bias,
                                                  float* __restrict__ out) {
  const int w = threadIdx.x >> 5;
  const int lane = threadIdx.x & 31;
  const int hf = lane >> 4;
  const int cl = lane & 15;
  const int m0 = blockIdx.y * 64 + w * 16;
  const int n0 = blockIdx.x * 64;
  v8f acc[4];
#pragma unroll
  for (int i = 0; i < 4; ++i)
#pragma unroll
    for (int r = 0; r < 8; ++r) acc[i][r] = 0.f;

  gemm_wave_16x64(AO, Wt, CEMB, m0, n0, acc);

#pragma unroll
  for (int i = 0; i < 4; ++i) {
    const int gn = n0 + i * 16 + cl;
    const float bv = bias[gn];
#pragma unroll
    for (int r = 0; r < 8; ++r) {
      const int gm = m0 + r + 8 * hf;
      out[(size_t)gm * CEMB + gn] = acc[i][r] + bv;
    }
  }
}

extern "C" void kernel_launch(void* const* d_in, const int* in_sizes, int n_in,
                              void* d_out, int out_size, void* d_ws, size_t ws_size,
                              hipStream_t stream) {
  (void)in_sizes; (void)n_in; (void)out_size; (void)ws_size;
  const float* x     = (const float*)d_in[0];
  const float* W_qkv = (const float*)d_in[1];
  const float* b_qkv = (const float*)d_in[2];
  const float* W_out = (const float*)d_in[3];
  const float* b_out = (const float*)d_in[4];
  float* out = (float*)d_out;

  // workspace carve-out (~88 MB total, bf16 staging)
  char* ws = (char*)d_ws;
  __bf16* xb  = (__bf16*)ws; ws += (size_t)MROWS * CEMB * 2;        // 16 MB
  __bf16* Wqt = (__bf16*)ws; ws += (size_t)3 * CEMB * CEMB * 2;     //  6 MB  (3072x1024)
  __bf16* Wot = (__bf16*)ws; ws += (size_t)CEMB * CEMB * 2;         //  2 MB
  __bf16* Qb  = (__bf16*)ws; ws += (size_t)NBH * TSEQ * HDIM * 2;   // 16 MB
  __bf16* Kb  = (__bf16*)ws; ws += (size_t)NBH * TSEQ * HDIM * 2;   // 16 MB
  __bf16* Vtb = (__bf16*)ws; ws += (size_t)NBH * TSEQ * HDIM * 2;   // 16 MB
  __bf16* AO  = (__bf16*)ws;                                        // 16 MB

  k_cvt<<<2048, 256, 0, stream>>>(x, xb, MROWS * CEMB);
  k_cvt_t<<<2048, 256, 0, stream>>>(W_qkv, Wqt, CEMB, 3 * CEMB);
  k_cvt_t<<<1024, 256, 0, stream>>>(W_out, Wot, CEMB, CEMB);

  dim3 g1(3 * CEMB / 64, MROWS / 64);          // (48,128)
  k_qkv_gemm<<<g1, 128, 0, stream>>>(xb, Wqt, b_qkv, Qb, Kb, Vtb);

  k_attn<<<(NBH * TSEQ / 16) / 4, 128, 0, stream>>>(Qb, Kb, Vtb, AO);

  dim3 g2(CEMB / 64, MROWS / 64);              // (16,128)
  k_out_gemm<<<g2, 128, 0, stream>>>(AO, Wot, b_out, out);
}
